// NLP_double_attention_27006754357593
// MI455X (gfx1250) — compile-verified, matrix operand order checked by reading, not run on hardware
//
#include <hip/hip_runtime.h>
#include <math.h>

#define R1   8
#define HID  256
#define LWD  256
#define NEVN 2048
#define WORD 32
#define DA   128
#define SQ   2050       // S = NEV + 2
#define SQP  2064       // S padded to multiple of 16

typedef __attribute__((ext_vector_type(16))) _Float16 v16h;
typedef __attribute__((ext_vector_type(8)))  _Float16 v8h;
typedef __attribute__((ext_vector_type(4)))  _Float16 v4h;
typedef __attribute__((ext_vector_type(2)))  _Float16 v2h;
typedef __attribute__((ext_vector_type(8)))  float    v8f;

__device__ __forceinline__ float sigf(float x){ return 1.0f / (1.0f + __expf(-x)); }

// ---- fragment loaders (f16 sources, guard-free, clause-friendly) -------------
// A fragment from row-major f16 A[*,K]: two contiguous 16B loads.
__device__ __forceinline__ v16h frag_a_h(const _Float16* A, int K, int row, int kc, int half){
  const _Float16* p = A + (size_t)row * K + kc * 32;
  v8h lo = *(const v8h*)(p + (half ? 8  : 0));
  v8h hi = *(const v8h*)(p + (half ? 24 : 16));
  v16h a;
#pragma unroll
  for (int i = 0; i < 8; i++){ a[i] = lo[i]; a[8 + i] = hi[i]; }
  return a;
}

// B fragment from pre-swizzled fragment-major layout: one contiguous 32B load.
__device__ __forceinline__ v16h frag_b_sw(const _Float16* Wf, int fragIdx, int lane){
  return *(const v16h*)(Wf + ((size_t)fragIdx * 32 + lane) * 16);
}

// ---- pre-swizzle weights: W[Nsrc,Ksrc] f32 -> fragment-major f16 -------------
// Layout: Wf[(kc*ntiles + nt)*512 + lane*16 + i]; zero-pads n>=Nsrc, k>=Ksrc.
__global__ void swizzle_w(const float* __restrict__ W, _Float16* __restrict__ Wf,
                          int Nsrc, int Ksrc, int ntiles, int kchunks){
  int idx = blockIdx.x * blockDim.x + threadIdx.x;
  if (idx >= kchunks * ntiles * 512) return;
  int i    = idx & 15;
  int lane = (idx >> 4) & 31;
  int frag = idx >> 9;
  int nt = frag % ntiles;
  int kc = frag / ntiles;
  int half = lane >> 4;
  int n = nt * 16 + (lane & 15);
  int koff = (i < 8) ? ((half ? 8 : 0) + i) : ((half ? 24 : 16) + (i - 8));
  int k = kc * 32 + koff;
  _Float16 v = (_Float16)0.0f;
  if (n < Nsrc && k < Ksrc) v = (_Float16)W[(size_t)n * Ksrc + k];
  Wf[idx] = v;
}

// ---- f32 -> f16 row-major conversion (4 elements/thread) ---------------------
__global__ void conv_f32_f16(const float* __restrict__ src, _Float16* __restrict__ dst, int count4){
  int i = blockIdx.x * blockDim.x + threadIdx.x;
  if (i >= count4) return;
  float4 v = ((const float4*)src)[i];
  v4h h; h[0]=(_Float16)v.x; h[1]=(_Float16)v.y; h[2]=(_Float16)v.z; h[3]=(_Float16)v.w;
  ((v4h*)dst)[i] = h;
}

__global__ void zero_tail_f32(float* __restrict__ p, int count){
  int i = blockIdx.x * blockDim.x + threadIdx.x;
  if (i < count) p[i] = 0.0f;
}
__global__ void zero_tail_f16(_Float16* __restrict__ p, int count){
  int i = blockIdx.x * blockDim.x + threadIdx.x;
  if (i < count) p[i] = (_Float16)0.0f;
}

// ---- Fused attention scores: S[M,8] = relu(A[M,K] @ W1^T + b1) @ W2^T + b2 ---
// A: f16 row-major; W1f: swizzled (ntiles=8, kchunks=K/32); W2f: swizzled (ntiles=1, kchunks=4).
__global__ void attn_score_fused(const _Float16* __restrict__ X, const _Float16* __restrict__ W1f,
                                 const float* __restrict__ b1,  const _Float16* __restrict__ W2f,
                                 const float* __restrict__ b2,  float* __restrict__ Sout,
                                 int M, int K){
  __shared__ _Float16 hbuf[8][DA][16];        // [wave][col][row], 32 KB
  int lane = threadIdx.x & 31;
  int wave = threadIdx.x >> 5;
  int tile = blockIdx.x * 8 + wave;
  int mtiles = M >> 4;
  if (tile >= mtiles) return;                 // wave-uniform
  int m0 = tile << 4;
  int cl = lane & 15, half = lane >> 4;

  v8f acc[8];
#pragma unroll
  for (int g = 0; g < 8; g++)
#pragma unroll
    for (int j = 0; j < 8; j++) acc[g][j] = 0.0f;

  int kchunks = K >> 5;
  for (int kc = 0; kc < kchunks; kc++){
    v16h a = frag_a_h(X, K, m0 + cl, kc, half);
#pragma unroll
    for (int g = 0; g < 8; g++){
      v16h b = frag_b_sw(W1f, kc * 8 + g, lane);
      acc[g] = __builtin_amdgcn_wmma_f32_16x16x32_f16(false, a, false, b, (short)0,
                                                      acc[g], false, false);
    }
  }

  // relu + bias -> LDS as f16 in [col][row] layout (paired 32-bit stores)
  int rbase = (lane < 16) ? 0 : 8;
#pragma unroll
  for (int g = 0; g < 8; g++){
    int col = g * 16 + cl;
    float bias = b1[col];
    v2h* dst = (v2h*)&hbuf[wave][col][rbase];
#pragma unroll
    for (int j = 0; j < 4; j++){
      float v0 = acc[g][2*j]   + bias;
      float v1 = acc[g][2*j+1] + bias;
      v2h pr; pr[0] = (_Float16)(v0 > 0.0f ? v0 : 0.0f);
              pr[1] = (_Float16)(v1 > 0.0f ? v1 : 0.0f);
      dst[j] = pr;
    }
  }

  // GEMM2: s = h @ W2^T  (K2 = 128 -> 4 chunks; N = 8 padded to 16 in W2f)
  v8f acc2;
#pragma unroll
  for (int j = 0; j < 8; j++) acc2[j] = 0.0f;
#pragma unroll
  for (int kk = 0; kk < 4; kk++){
    int kb0 = kk * 32 + (half ? 8  : 0);
    int kb1 = kk * 32 + (half ? 24 : 16);
    v16h a2;
#pragma unroll
    for (int i = 0; i < 8; i++){
      a2[i]     = hbuf[wave][kb0 + i][cl];
      a2[8 + i] = hbuf[wave][kb1 + i][cl];
    }
    v16h b2v = frag_b_sw(W2f, kk, lane);
    acc2 = __builtin_amdgcn_wmma_f32_16x16x32_f16(false, a2, false, b2v, (short)0,
                                                  acc2, false, false);
  }
  if (cl < R1){
    float bv = b2[cl];
#pragma unroll
    for (int j = 0; j < 8; j++){
      int row = m0 + rbase + j;
      Sout[(size_t)row * R1 + cl] = acc2[j] + bv;
    }
  }
}

// ---- Per-n masked softmax over l + weighted pool: Out[n,32] -------------------
__global__ void attn_pool_finish(const float* __restrict__ Sbuf, const float* __restrict__ X,
                                 const int* __restrict__ lens, float* __restrict__ Out){
  __shared__ float red[256];
  __shared__ float wgt[256];
  __shared__ float part[8][32];
  int n = blockIdx.x;
  int l = threadIdx.x;                         // L == 256 == blockDim
  int len = lens[n];
  bool valid = l < len;
  const float* srow = Sbuf + ((size_t)l * NEVN + n) * R1;
  float sv[R1];
#pragma unroll
  for (int r = 0; r < R1; r++) sv[r] = valid ? srow[r] : -1e30f;

  float w = 0.0f;
  for (int r = 0; r < R1; r++){
    red[l] = sv[r];  __syncthreads();
    for (int off = 128; off > 0; off >>= 1){
      if (l < off) red[l] = fmaxf(red[l], red[l + off]);
      __syncthreads();
    }
    float mx = red[0]; __syncthreads();
    float e = valid ? __expf(sv[r] - mx) : 0.0f;
    red[l] = e; __syncthreads();
    for (int off = 128; off > 0; off >>= 1){
      if (l < off) red[l] += red[l + off];
      __syncthreads();
    }
    float sm = red[0]; __syncthreads();
    w += e / sm;
  }
  wgt[l] = w * (1.0f / R1);
  __syncthreads();

  int d = l & 31, c = l >> 5;                  // 8 chunks of 32 rows
  float p = 0.0f;
  for (int j = 0; j < 32; j++){
    int ll = c * 32 + j;
    p += wgt[ll] * X[((size_t)ll * NEVN + n) * WORD + d];
  }
  part[c][d] = p;
  __syncthreads();
  if (l < 32){
    float s = 0.0f;
#pragma unroll
    for (int cc = 0; cc < 8; cc++) s += part[cc][l];
    Out[(size_t)n * WORD + l] = s;
  }
}

// ---- event[n,0..127] (f16) = concat(inputs[1+n,:50], avg_sent[n], avg_it2[n], 0)
__global__ void build_event(const float* __restrict__ inputs, const float* __restrict__ avg_sent,
                            const float* __restrict__ avg_it2, _Float16* __restrict__ Ev){
  int idx = blockIdx.x * blockDim.x + threadIdx.x;
  if (idx >= NEVN * DA) return;
  int n = idx / DA, k = idx % DA;
  float v = 0.0f;
  if (k < 50)       v = inputs[(size_t)(1 + n) * 83 + k];
  else if (k < 82)  v = avg_sent[n * 32 + (k - 50)];
  else if (k < 114) v = avg_it2[n * 32 + (k - 82)];
  Ev[idx] = (_Float16)v;
}

// ---- Generic WMMA GEMM: C[M,N] = A[M,K] @ W^T + bias0 (+ bias1) --------------
// A: f16 row-major, M mult of 16, K mult of 32, N mult of 16; Wf swizzled.
__global__ void gemm_bias(const _Float16* __restrict__ A, const _Float16* __restrict__ Wf,
                          const float* __restrict__ bias0, const float* __restrict__ bias1,
                          float* __restrict__ C, int M, int K, int N){
  int lane = threadIdx.x & 31, wave = threadIdx.x >> 5;
  int mtiles = M >> 4, ntiles = N >> 4;
  int tile = blockIdx.x * 8 + wave;
  if (tile >= mtiles * ntiles) return;
  int mt = tile / ntiles, nt = tile % ntiles;
  int m0 = mt << 4, n0 = nt << 4;
  int cl = lane & 15, half = lane >> 4;

  v8f acc;
#pragma unroll
  for (int j = 0; j < 8; j++) acc[j] = 0.0f;

  int kchunks = K >> 5;
  for (int kc = 0; kc < kchunks; kc++){
    v16h a = frag_a_h(A, K, m0 + cl, kc, half);
    v16h b = frag_b_sw(Wf, kc * ntiles + nt, lane);
    acc = __builtin_amdgcn_wmma_f32_16x16x32_f16(false, a, false, b, (short)0,
                                                 acc, false, false);
  }
  int rbase = (lane < 16) ? 0 : 8;
  int col = n0 + cl;
  float bs = bias0[col] + (bias1 ? bias1[col] : 0.0f);
#pragma unroll
  for (int j = 0; j < 8; j++){
    int row = m0 + rbase + j;
    C[(size_t)row * N + col] = acc[j] + bs;
  }
}

// ---- begin/end edge rows ------------------------------------------------------
__global__ void begin_end(const float* __restrict__ inputs, const float* __restrict__ w_edge,
                          const float* __restrict__ b_edge, float* __restrict__ out0){
  int t = threadIdx.x;
  const float* src = (t < 128) ? inputs : inputs + (size_t)(SQ - 1) * 83;
  int i = t & 127;
  float acc = b_edge[i];
  const float* wr = w_edge + (size_t)i * 83;
  for (int k = 0; k < 83; k++) acc += wr[k] * src[k];
  if (t < 128) out0[i] = acc;
  else         out0[(size_t)(SQ - 1) * DA + i] = acc;
}

// ---- LSTM recurrence (one direction); writes f16 hidden (+optional f32) ------
__global__ void __launch_bounds__(1024)
lstm_recur(const float* __restrict__ xproj,   // [SQP,1024] = x@Wih^T + bih + bhh
           const float* __restrict__ Whh,     // [1024,256]
           _Float16* __restrict__ outh,       // f16 h at outh[t*512 + i]
           float* __restrict__ outp,          // optional f32 h (may be null)
           int reverse){
  __shared__ __align__(16) float hs[HID];
  __shared__ float cs[HID];
  __shared__ float gs[4 * HID];
  int tid = threadIdx.x;
  if (tid < HID){ hs[tid] = 0.0f; cs[tid] = 0.0f; }
  __syncthreads();
  const float4* wr = (const float4*)(Whh + (size_t)tid * HID);
  for (int step = 0; step < SQ; step++){
    int t = reverse ? (SQ - 1 - step) : step;
    float acc = xproj[(size_t)t * 1024 + tid];
    const float4* h4 = (const float4*)hs;
#pragma unroll 8
    for (int j = 0; j < HID / 4; j++){
      float4 w = wr[j]; float4 h = h4[j];
      acc += w.x * h.x + w.y * h.y + w.z * h.z + w.w * h.w;
    }
    gs[tid] = acc;
    __syncthreads();
    if (tid < HID){
      float ig = sigf(gs[tid]);
      float fg = sigf(gs[HID + tid]);
      float gg = tanhf(gs[2 * HID + tid]);
      float og = sigf(gs[3 * HID + tid]);
      float c  = fg * cs[tid] + ig * gg;
      float h  = og * tanhf(c);
      cs[tid] = c; hs[tid] = h;
      outh[(size_t)t * 512 + tid] = (_Float16)h;
      if (outp) outp[(size_t)t * 512 + tid] = h;
    }
    __syncthreads();
  }
}

// ---- Final: softmax pool over sequence + 2-layer head -> scalar ---------------
__global__ void final_head(const float* __restrict__ Sev, const float* __restrict__ out2,
                           const int* __restrict__ seq_len,
                           const float* __restrict__ w_inter, const float* __restrict__ b_inter,
                           const float* __restrict__ w_fin,   const float* __restrict__ b_fin,
                           float* __restrict__ out){
  __shared__ float red[256];
  __shared__ float wl[SQ];
  __shared__ float avg[512];
  __shared__ float mx[R1], sm[R1];
  __shared__ float inter[32];
  int tid = threadIdx.x;
  int Sv = seq_len[0]; if (Sv > SQ) Sv = SQ;

  for (int r = 0; r < R1; r++){
    float m = -1e30f;
    for (int t = tid; t < Sv; t += 256) m = fmaxf(m, Sev[(size_t)t * R1 + r]);
    red[tid] = m; __syncthreads();
    for (int off = 128; off > 0; off >>= 1){
      if (tid < off) red[tid] = fmaxf(red[tid], red[tid + off]);
      __syncthreads();
    }
    if (tid == 0) mx[r] = red[0];
    __syncthreads();
    float s = 0.0f;
    for (int t = tid; t < Sv; t += 256) s += __expf(Sev[(size_t)t * R1 + r] - mx[r]);
    red[tid] = s; __syncthreads();
    for (int off = 128; off > 0; off >>= 1){
      if (tid < off) red[tid] += red[tid + off];
      __syncthreads();
    }
    if (tid == 0) sm[r] = red[0];
    __syncthreads();
  }
  for (int t = tid; t < SQ; t += 256){
    float w = 0.0f;
    if (t < Sv)
      for (int r = 0; r < R1; r++) w += __expf(Sev[(size_t)t * R1 + r] - mx[r]) / sm[r];
    wl[t] = w * (1.0f / R1);
  }
  __syncthreads();
  for (int d = tid; d < 512; d += 256){
    float a = 0.0f;
    for (int t = 0; t < Sv; t++) a += wl[t] * out2[(size_t)t * 512 + d];
    avg[d] = a;
  }
  __syncthreads();
  if (tid < 32){
    float v = b_inter[tid];
    const float* wr = w_inter + (size_t)tid * 512;
    for (int d = 0; d < 512; d++) v += wr[d] * avg[d];
    inter[tid] = v > 0.0f ? v : 0.0f;
  }
  __syncthreads();
  if (tid == 0){
    float o = b_fin[0];
    for (int i = 0; i < 32; i++) o += inter[i] * w_fin[i];
    out[0] = o;
  }
}

extern "C" void kernel_launch(void* const* d_in, const int* in_sizes, int n_in,
                              void* d_out, int out_size, void* d_ws, size_t ws_size,
                              hipStream_t stream) {
  (void)in_sizes; (void)n_in; (void)out_size; (void)ws_size;
  const float* item      = (const float*)d_in[0];
  const float* item2     = (const float*)d_in[1];
  const float* inputs    = (const float*)d_in[2];
  const int*   item_len  = (const int*)d_in[3];
  const int*   item2_len = (const int*)d_in[4];
  const int*   seq_len   = (const int*)d_in[5];
  const float* w1_item  = (const float*)d_in[6],  *b1_item  = (const float*)d_in[7];
  const float* w2_item  = (const float*)d_in[8],  *b2_item  = (const float*)d_in[9];
  const float* w1_item2 = (const float*)d_in[10], *b1_item2 = (const float*)d_in[11];
  const float* w2_item2 = (const float*)d_in[12], *b2_item2 = (const float*)d_in[13];
  const float* w_evt    = (const float*)d_in[14], *b_evt    = (const float*)d_in[15];
  const float* w_edge   = (const float*)d_in[16], *b_edge   = (const float*)d_in[17];
  const float* w1_ev    = (const float*)d_in[18], *b1_ev    = (const float*)d_in[19];
  const float* w2_ev    = (const float*)d_in[20], *b2_ev    = (const float*)d_in[21];
  const float* w_inter  = (const float*)d_in[22], *b_inter  = (const float*)d_in[23];
  const float* w_fin    = (const float*)d_in[24], *b_fin    = (const float*)d_in[25];
  const float* Wih_f0 = (const float*)d_in[26], *Whh_f0 = (const float*)d_in[27];
  const float* bih_f0 = (const float*)d_in[28], *bhh_f0 = (const float*)d_in[29];
  const float* Wih_b0 = (const float*)d_in[30], *Whh_b0 = (const float*)d_in[31];
  const float* bih_b0 = (const float*)d_in[32], *bhh_b0 = (const float*)d_in[33];
  const float* Wih_f1 = (const float*)d_in[34], *Whh_f1 = (const float*)d_in[35];
  const float* bih_f1 = (const float*)d_in[36], *bhh_f1 = (const float*)d_in[37];
  const float* Wih_b1 = (const float*)d_in[38], *Whh_b1 = (const float*)d_in[39];
  const float* bih_b1 = (const float*)d_in[40], *bhh_b1 = (const float*)d_in[41];
  float* out = (float*)d_out;

  // ---- workspace layout: f32 region then f16 region ----
  float* ws = (float*)d_ws;
  size_t o = 0;
  float* s_buf    = ws + o; o += (size_t)LWD * NEVN * R1;   // 4,194,304
  float* avg_sent = ws + o; o += (size_t)NEVN * WORD;
  float* avg_it2  = ws + o; o += (size_t)NEVN * WORD;
  float* out0     = ws + o; o += (size_t)SQP * DA;
  float* xpf      = ws + o; o += (size_t)SQP * 1024;
  float* xpb      = ws + o; o += (size_t)SQP * 1024;
  float* out2     = ws + o; o += (size_t)SQP * 512;
  float* s_ev     = ws + o; o += (size_t)SQP * R1;

  _Float16* hbase = (_Float16*)(ws + o);
  size_t h = 0;
  _Float16* item16  = hbase + h; h += (size_t)LWD * NEVN * WORD;  // 16,777,216 halves
  _Float16* eventh  = hbase + h; h += (size_t)NEVN * DA;
  _Float16* out0h   = hbase + h; h += (size_t)SQP * DA;
  _Float16* out1h   = hbase + h; h += (size_t)SQP * 512;
  _Float16* out2h   = hbase + h; h += (size_t)SQP * 512;
  _Float16* w1f_a   = hbase + h; h += 1  * 8  * 512;   // w1_item  [128,32]
  _Float16* w2f_a   = hbase + h; h += 4  * 1  * 512;   // w2_item  [8,128]
  _Float16* w1f_b   = hbase + h; h += 1  * 8  * 512;
  _Float16* w2f_b   = hbase + h; h += 4  * 1  * 512;
  _Float16* wevtf   = hbase + h; h += 4  * 8  * 512;   // w_evt [128,114] -> K pad 128
  _Float16* w1evf   = hbase + h; h += 16 * 8  * 512;   // w1_ev [128,512]
  _Float16* w2evf   = hbase + h; h += 4  * 1  * 512;
  _Float16* wf_f0   = hbase + h; h += 4  * 64 * 512;   // Wih_f0 [1024,128]
  _Float16* wf_b0   = hbase + h; h += 4  * 64 * 512;
  _Float16* wf_f1   = hbase + h; h += 16 * 64 * 512;   // Wih_f1 [1024,512]
  _Float16* wf_b1   = hbase + h; h += 16 * 64 * 512;

  // ---- pre-swizzle all GEMM weights into fragment-major f16 ----
  swizzle_w<<<(1*8*512 + 255)/256, 256, 0, stream>>>(w1_item,  w1f_a, 128,  32, 8, 1);
  swizzle_w<<<(4*1*512 + 255)/256, 256, 0, stream>>>(w2_item,  w2f_a, 8,   128, 1, 4);
  swizzle_w<<<(1*8*512 + 255)/256, 256, 0, stream>>>(w1_item2, w1f_b, 128,  32, 8, 1);
  swizzle_w<<<(4*1*512 + 255)/256, 256, 0, stream>>>(w2_item2, w2f_b, 8,   128, 1, 4);
  swizzle_w<<<(4*8*512 + 255)/256, 256, 0, stream>>>(w_evt,    wevtf, 128, 114, 8, 4);
  swizzle_w<<<(16*8*512 + 255)/256, 256, 0, stream>>>(w1_ev,   w1evf, 128, 512, 8, 16);
  swizzle_w<<<(4*1*512 + 255)/256, 256, 0, stream>>>(w2_ev,    w2evf, 8,   128, 1, 4);
  swizzle_w<<<(4*64*512 + 255)/256, 256, 0, stream>>>(Wih_f0,  wf_f0, 1024, 128, 64, 4);
  swizzle_w<<<(4*64*512 + 255)/256, 256, 0, stream>>>(Wih_b0,  wf_b0, 1024, 128, 64, 4);
  swizzle_w<<<(16*64*512 + 255)/256, 256, 0, stream>>>(Wih_f1, wf_f1, 1024, 512, 64, 16);
  swizzle_w<<<(16*64*512 + 255)/256, 256, 0, stream>>>(Wih_b1, wf_b1, 1024, 512, 64, 16);

  // ---- zero pad rows that feed GEMM-A reads ----
  zero_tail_f32<<<(((SQP-SQ)*DA)  + 255)/256, 256, 0, stream>>>(out0  + (size_t)SQ * DA,  (SQP-SQ) * DA);
  zero_tail_f16<<<(((SQP-SQ)*512) + 255)/256, 256, 0, stream>>>(out1h + (size_t)SQ * 512, (SQP-SQ) * 512);
  zero_tail_f16<<<(((SQP-SQ)*512) + 255)/256, 256, 0, stream>>>(out2h + (size_t)SQ * 512, (SQP-SQ) * 512);

  const int M_A = LWD * NEVN;                 // 524288 rows, 32768 tiles
  const int C4_ITEM = M_A * WORD / 4;
  // --- item attention pooling ---
  conv_f32_f16<<<(C4_ITEM + 255)/256, 256, 0, stream>>>(item, item16, C4_ITEM);
  attn_score_fused<<<4096, 256, 0, stream>>>(item16, w1f_a, b1_item, w2f_a, b2_item, s_buf, M_A, WORD);
  attn_pool_finish<<<NEVN, 256, 0, stream>>>(s_buf, item,  item_len,  avg_sent);
  // --- item2 attention pooling (reuse item16/s_buf) ---
  conv_f32_f16<<<(C4_ITEM + 255)/256, 256, 0, stream>>>(item2, item16, C4_ITEM);
  attn_score_fused<<<4096, 256, 0, stream>>>(item16, w1f_b, b1_item2, w2f_b, b2_item2, s_buf, M_A, WORD);
  attn_pool_finish<<<NEVN, 256, 0, stream>>>(s_buf, item2, item2_len, avg_it2);

  // --- event features + embedding + edge rows -> out0 [S,128] -> f16 ---
  build_event<<<(NEVN * DA + 255)/256, 256, 0, stream>>>(inputs, avg_sent, avg_it2, eventh);
  gemm_bias<<<128, 256, 0, stream>>>(eventh, wevtf, b_evt, nullptr, out0 + DA, NEVN, DA, DA);
  begin_end<<<1, 256, 0, stream>>>(inputs, w_edge, b_edge, out0);
  conv_f32_f16<<<((SQP*DA/4) + 255)/256, 256, 0, stream>>>(out0, out0h, SQP * DA / 4);

  // --- BLSTM layer 0 ---  (M=2064 -> 129*64 = 8256 tiles -> 1032 blocks)
  gemm_bias<<<1032, 256, 0, stream>>>(out0h, wf_f0, bih_f0, bhh_f0, xpf, SQP, DA, 1024);
  gemm_bias<<<1032, 256, 0, stream>>>(out0h, wf_b0, bih_b0, bhh_b0, xpb, SQP, DA, 1024);
  lstm_recur<<<1, 1024, 0, stream>>>(xpf, Whh_f0, out1h,       nullptr, 0);
  lstm_recur<<<1, 1024, 0, stream>>>(xpb, Whh_b0, out1h + HID, nullptr, 1);

  // --- BLSTM layer 1 ---
  gemm_bias<<<1032, 256, 0, stream>>>(out1h, wf_f1, bih_f1, bhh_f1, xpf, SQP, 512, 1024);
  gemm_bias<<<1032, 256, 0, stream>>>(out1h, wf_b1, bih_b1, bhh_b1, xpb, SQP, 512, 1024);
  lstm_recur<<<1, 1024, 0, stream>>>(xpf, Whh_f1, out2h,       out2,       0);
  lstm_recur<<<1, 1024, 0, stream>>>(xpb, Whh_b1, out2h + HID, out2 + HID, 1);

  // --- final attention over sequence + head ---  (129 tiles -> 17 blocks)
  attn_score_fused<<<17, 256, 0, stream>>>(out2h, w1evf, b1_ev, w2evf, b2_ev, s_ev, SQP, 512);
  final_head<<<1, 256, 0, stream>>>(s_ev, out2, seq_len, w_inter, b_inter, w_fin, b_fin, out);
}
